// SequenceModel_15487652069814
// MI455X (gfx1250) — matrix operand order repaired and checked by reading, not compiled, not run on hardware
//
#include <hip/hip_runtime.h>
#include <hip/hip_bf16.h>

typedef __bf16 bf16;
typedef __attribute__((ext_vector_type(16))) __bf16 v16bf;
typedef __attribute__((ext_vector_type(8)))  __bf16 v8bf;
typedef __attribute__((ext_vector_type(8)))  float  v8f;

#define TSTEPS 32
#define BATCH  2048
#define DETER  2048
#define HID    1024
#define GATES  6144           // 3*DETER
#define INK    1088           // 1072 padded to 34*32
#define BM 128
#define BN 128
#define BKS 32
#define BKP 40                // padded LDS row stride (bf16) -> 80B rows, conflict-friendly

// ---------------------------------------------------------------------------
// CDNA5 async global->LDS path (ASYNCcnt), with guarded fallback to sync stage
// ---------------------------------------------------------------------------
#if defined(__AMDGCN__) && __has_builtin(__builtin_amdgcn_global_load_async_to_lds_b128)
#define HAVE_ASYNC_LDS 1
#else
#define HAVE_ASYNC_LDS 0
#endif

#if HAVE_ASYNC_LDS
#define AS1 __attribute__((address_space(1)))
#define AS3 __attribute__((address_space(3)))
typedef int v4i __attribute__((vector_size(16)));   // b128 payload type of the builtin
__device__ __forceinline__ AS1 v4i* glb_cast(const void* p) {
  return (AS1 v4i*)(uintptr_t)p;
}
__device__ __forceinline__ AS3 v4i* lds_cast(void* p) {
  // generic LDS address: low 32 bits are the raw LDS offset (ISA aperture rule)
  return (AS3 v4i*)(unsigned)(uintptr_t)p;
}
#if __has_builtin(__builtin_amdgcn_s_wait_asynccnt)
#define ASYNC_WAIT(n) __builtin_amdgcn_s_wait_asynccnt(n)
#else
#define ASYNC_WAIT(n) asm volatile("s_wait_asynccnt %0" :: "n"(n))
#endif
#else
#define ASYNC_WAIT(n) do {} while (0)
#endif

// ---------------------------------------------------------------------------
// CDNA5 Tensor Data Mover (TDM) path for weight tiles (TENSORcnt)
// ---------------------------------------------------------------------------
#if HAVE_ASYNC_LDS && __has_builtin(__builtin_amdgcn_tensor_load_to_lds) && \
    __has_builtin(__builtin_amdgcn_s_wait_tensorcnt)
#define HAVE_TDM 1
#else
#define HAVE_TDM 0
#endif

#if HAVE_TDM
typedef unsigned int u32x4 __attribute__((vector_size(16)));
typedef int          i32x8 __attribute__((vector_size(32)));
typedef int          i32x4 __attribute__((vector_size(16)));
#define TENSOR_WAIT(n) __builtin_amdgcn_s_wait_tensorcnt(n)
// DMA one BN x BKS bf16 tile (row stride ldw elems) into LDS with 16B row pad.
__device__ __forceinline__ void tdm_load_tile(const bf16* gsrc, unsigned lds_off,
                                              unsigned ldw_elems) {
  const unsigned long long ga = (unsigned long long)(uintptr_t)gsrc;
  // D# group0: count=1 | lds_addr | global_addr[56:0] | type=2
  u32x4 g0 = { 1u, lds_off, (unsigned)ga,
               (unsigned)((ga >> 32) & 0x1FFFFFFu) | 0x80000000u };
  // D# group1: data_size=2B, pad_enable, pad_interval=16 DW (64B), pad_amount=4 DW (16B)
  const unsigned d0  = (1u << 16) | (1u << 20) | (3u << 22) | (3u << 25);
  const unsigned td0 = ldw_elems;      // tensor dim0 (elements)  -> no OOB for our tiles
  const unsigned td1 = 1u << 20;       // tensor dim1 (rows)      -> no OOB
  i32x8 g1 = { (int)d0,
               (int)((td0 & 0xFFFFu) << 16),
               (int)(((td0 >> 16) & 0xFFFFu) | ((td1 & 0xFFFFu) << 16)),
               (int)(((td1 >> 16) & 0xFFFFu) | ((unsigned)BKS << 16)),  // tile_dim0=32
               (int)(unsigned)BN,                                       // tile_dim1=128
               (int)(unsigned)ldw_elems,                                // dim0 stride lo32
               0, 0 };
  i32x4 gz  = { 0, 0, 0, 0 };          // groups 2/3 unused (2D tensor)
  i32x8 gz8 = { 0, 0, 0, 0, 0, 0, 0, 0 };
  __builtin_amdgcn_tensor_load_to_lds(g0, g1, gz, gz, gz8, 0);
}
#define A_OPS 2
#else
#define TENSOR_WAIT(n) do {} while (0)
#define A_OPS 4
#endif

__device__ __forceinline__ v16bf cat8(v8bf a, v8bf b) {
  return __builtin_shufflevector(a, b, 0,1,2,3,4,5,6,7,8,9,10,11,12,13,14,15);
}

// ---------------------------------------------------------------------------
// Generic bf16 WMMA GEMM:  C[M,N] = A[M,K] * W[N,K]^T (+ bias[N])
// Block: 256 threads = 8 waves; block tile 128x128; wave tile 64x32 (4x2 WMMA)
// Double-buffered LDS; A streamed with async-to-LDS, W streamed with TDM DMA.
// ---------------------------------------------------------------------------
__global__ __launch_bounds__(256) void wmma_gemm_bias_kernel(
    const bf16* __restrict__ A, int lda,
    const bf16* __restrict__ W, int ldw,
    const float* __restrict__ bias,
    float* __restrict__ C, int ldc, int K)
{
  __shared__ bf16 ldsA[2][BM * BKP];
  __shared__ bf16 ldsB[2][BN * BKP];
  const int tid  = threadIdx.x;
  const int lane = tid & 31;
  const int wave = tid >> 5;
  const int wm   = (wave >> 2) * 64;   // 0 or 64
  const int wn   = (wave & 3) * 32;    // 0,32,64,96
  const int kh   = lane >> 4;          // lane half (K-half selector)
  const int l16  = lane & 15;
  const size_t bm0 = (size_t)blockIdx.y * BM;
  const size_t bn0 = (size_t)blockIdx.x * BN;

  v8f acc[4][2];
#pragma unroll
  for (int i = 0; i < 4; ++i)
#pragma unroll
    for (int j = 0; j < 2; ++j) acc[i][j] = v8f{};

  const int r = tid >> 1;          // staging row 0..127
  const int c = (tid & 1) << 4;    // staging col 0 or 16
  const bf16* gA = A + (bm0 + r) * (size_t)lda + c;
  const bf16* gB = W + (bn0 + r) * (size_t)ldw + c;
  const bf16* gBtile = W + bn0 * (size_t)ldw;   // tile row 0 (for TDM)
  (void)gB; (void)gBtile;

  auto stage = [&](int buf, int k0) {
#if HAVE_TDM
    if (wave == 0)
      tdm_load_tile(gBtile + k0,
                    (unsigned)(uintptr_t)&ldsB[0][0] + (unsigned)buf * (BN * BKP * 2),
                    (unsigned)ldw);
    __builtin_amdgcn_global_load_async_to_lds_b128(
        glb_cast(gA + k0),     lds_cast(&ldsA[buf][r * BKP + c]),     0, 0);
    __builtin_amdgcn_global_load_async_to_lds_b128(
        glb_cast(gA + k0 + 8), lds_cast(&ldsA[buf][r * BKP + c + 8]), 0, 0);
#elif HAVE_ASYNC_LDS
    __builtin_amdgcn_global_load_async_to_lds_b128(
        glb_cast(gA + k0),     lds_cast(&ldsA[buf][r * BKP + c]),     0, 0);
    __builtin_amdgcn_global_load_async_to_lds_b128(
        glb_cast(gA + k0 + 8), lds_cast(&ldsA[buf][r * BKP + c + 8]), 0, 0);
    __builtin_amdgcn_global_load_async_to_lds_b128(
        glb_cast(gB + k0),     lds_cast(&ldsB[buf][r * BKP + c]),     0, 0);
    __builtin_amdgcn_global_load_async_to_lds_b128(
        glb_cast(gB + k0 + 8), lds_cast(&ldsB[buf][r * BKP + c + 8]), 0, 0);
#else
    *(v8bf*)&ldsA[buf][r * BKP + c]     = *(const v8bf*)(gA + k0);
    *(v8bf*)&ldsA[buf][r * BKP + c + 8] = *(const v8bf*)(gA + k0 + 8);
    *(v8bf*)&ldsB[buf][r * BKP + c]     = *(const v8bf*)(gB + k0);
    *(v8bf*)&ldsB[buf][r * BKP + c + 8] = *(const v8bf*)(gB + k0 + 8);
#endif
  };

  auto compute = [&](int cur) {
    v16bf bfrag[2], afrag[4];
#pragma unroll
    for (int j = 0; j < 2; ++j) {
      const int n = wn + j * 16 + l16;
      v8bf b0 = *(const v8bf*)&ldsB[cur][n * BKP + 16 * kh];
      v8bf b1 = *(const v8bf*)&ldsB[cur][n * BKP + 16 * kh + 8];
      bfrag[j] = cat8(b0, b1);
    }
#pragma unroll
    for (int i = 0; i < 4; ++i) {
      const int m = wm + i * 16 + l16;
      v8bf a0 = *(const v8bf*)&ldsA[cur][m * BKP + 16 * kh];
      v8bf a1 = *(const v8bf*)&ldsA[cur][m * BKP + 16 * kh + 8];
      afrag[i] = cat8(a0, a1);
    }
#pragma unroll
    for (int i = 0; i < 4; ++i)
#pragma unroll
      for (int j = 0; j < 2; ++j)
        acc[i][j] = __builtin_amdgcn_wmma_f32_16x16x32_bf16(
            false, afrag[i], false, bfrag[j], (short)0, acc[i][j], false, false);
  };

  stage(0, 0);
  const int nK = K / BKS;
  for (int kt = 0; kt < nK - 1; ++kt) {        // steady state: branch-free waits
    const int cur = kt & 1;
    stage(cur ^ 1, (kt + 1) * BKS);            // prefetch next tile
    TENSOR_WAIT(1);                            // current W-tile DMA complete
    ASYNC_WAIT(A_OPS);                         // current A-tile asyncs complete
    __syncthreads();
    compute(cur);
    __syncthreads();                           // done reading before overwrite
  }
  TENSOR_WAIT(0);
  ASYNC_WAIT(0);
  __syncthreads();
  compute((nK - 1) & 1);

#pragma unroll
  for (int i = 0; i < 4; ++i)
#pragma unroll
    for (int j = 0; j < 2; ++j) {
      const int n = (int)bn0 + wn + j * 16 + l16;
      const float bv = bias ? bias[n] : 0.0f;
#pragma unroll
      for (int e = 0; e < 8; ++e) {
        const size_t m = bm0 + wm + i * 16 + kh * 8 + e;
        C[m * (size_t)ldc + n] = acc[i][j][e] + bv;
      }
    }
}

// ---------------------------------------------------------------------------
// Phase-1 GEMM: A is the virtual f32 concat [stoch(1024) | actions(48) | pad]
// converted to bf16 during LDS staging (sync path). C = h_pre (f32).
// ---------------------------------------------------------------------------
__global__ __launch_bounds__(256) void wmma_gemm_concat_kernel(
    const float* __restrict__ stoch, const float* __restrict__ act,
    const bf16* __restrict__ W, int ldw,
    float* __restrict__ C, int ldc, int K)
{
  __shared__ bf16 ldsA[BM * BKP];
  __shared__ bf16 ldsB[BN * BKP];
  const int tid  = threadIdx.x;
  const int lane = tid & 31;
  const int wave = tid >> 5;
  const int wm   = (wave >> 2) * 64;
  const int wn   = (wave & 3) * 32;
  const int kh   = lane >> 4;
  const int l16  = lane & 15;
  const size_t bm0 = (size_t)blockIdx.y * BM;
  const size_t bn0 = (size_t)blockIdx.x * BN;

  v8f acc[4][2];
#pragma unroll
  for (int i = 0; i < 4; ++i)
#pragma unroll
    for (int j = 0; j < 2; ++j) acc[i][j] = v8f{};

  const int r = tid >> 1;
  const int c = (tid & 1) << 4;

  for (int k0 = 0; k0 < K; k0 += BKS) {
    const size_t R = bm0 + r;   // flattened t*B + b
#pragma unroll
    for (int e = 0; e < 16; ++e) {
      const int k = k0 + c + e;
      float v;
      if (k < 1024)      v = stoch[R * 1024 + k];
      else if (k < 1072) v = act[R * 48 + (k - 1024)];
      else               v = 0.0f;
      ldsA[r * BKP + c + e] = (bf16)v;
    }
    const bf16* sb = W + (bn0 + r) * (size_t)ldw + k0 + c;
    *(v8bf*)&ldsB[r * BKP + c]     = *(const v8bf*)(sb);
    *(v8bf*)&ldsB[r * BKP + c + 8] = *(const v8bf*)(sb + 8);
    __syncthreads();

    v16bf bfrag[2], afrag[4];
#pragma unroll
    for (int j = 0; j < 2; ++j) {
      const int n = wn + j * 16 + l16;
      v8bf b0 = *(const v8bf*)&ldsB[n * BKP + 16 * kh];
      v8bf b1 = *(const v8bf*)&ldsB[n * BKP + 16 * kh + 8];
      bfrag[j] = cat8(b0, b1);
    }
#pragma unroll
    for (int i = 0; i < 4; ++i) {
      const int m = wm + i * 16 + l16;
      v8bf a0 = *(const v8bf*)&ldsA[m * BKP + 16 * kh];
      v8bf a1 = *(const v8bf*)&ldsA[m * BKP + 16 * kh + 8];
      afrag[i] = cat8(a0, a1);
    }
#pragma unroll
    for (int i = 0; i < 4; ++i)
#pragma unroll
      for (int j = 0; j < 2; ++j)
        acc[i][j] = __builtin_amdgcn_wmma_f32_16x16x32_bf16(
            false, afrag[i], false, bfrag[j], (short)0, acc[i][j], false, false);
    __syncthreads();
  }

#pragma unroll
  for (int i = 0; i < 4; ++i)
#pragma unroll
    for (int j = 0; j < 2; ++j) {
      const int n = (int)bn0 + wn + j * 16 + l16;
#pragma unroll
      for (int e = 0; e < 8; ++e) {
        const size_t m = bm0 + wm + i * 16 + kh * 8 + e;
        C[m * (size_t)ldc + n] = acc[i][j][e];
      }
    }
}

// ---------------------------------------------------------------------------
// LayerNorm(eps=1e-3) + affine + SiLU, one block per row of 1024; bf16 out.
// ---------------------------------------------------------------------------
__global__ __launch_bounds__(256) void ln_silu_kernel(
    const float* __restrict__ Hpre, const float* __restrict__ gamma,
    const float* __restrict__ beta, bf16* __restrict__ feat)
{
  __shared__ float red[256];
  const size_t row = blockIdx.x;
  const int tid = threadIdx.x;
  const float* h = Hpre + row * HID;
  float v[4];
  float s = 0.0f;
#pragma unroll
  for (int i = 0; i < 4; ++i) { v[i] = h[tid + 256 * i]; s += v[i]; }
  red[tid] = s;
  __syncthreads();
  for (int o = 128; o > 0; o >>= 1) {
    if (tid < o) red[tid] += red[tid + o];
    __syncthreads();
  }
  const float mu = red[0] * (1.0f / HID);
  __syncthreads();
  float q = 0.0f;
#pragma unroll
  for (int i = 0; i < 4; ++i) { const float d = v[i] - mu; q += d * d; }
  red[tid] = q;
  __syncthreads();
  for (int o = 128; o > 0; o >>= 1) {
    if (tid < o) red[tid] += red[tid + o];
    __syncthreads();
  }
  const float rstd = rsqrtf(red[0] * (1.0f / HID) + 1e-3f);
#pragma unroll
  for (int i = 0; i < 4; ++i) {
    const int col = tid + 256 * i;
    const float g = (v[i] - mu) * rstd * gamma[col] + beta[col];
    const float f = g / (1.0f + __expf(-g));   // SiLU
    feat[row * HID + col] = (bf16)f;
  }
}

// ---------------------------------------------------------------------------
// GRU pointwise: r/z/n gates, h update; writes f32 out[t] + bf16 hidden mirror
// ---------------------------------------------------------------------------
__global__ __launch_bounds__(256) void gru_pointwise_kernel(
    const float* __restrict__ gi, const float* __restrict__ gh,
    const float* __restrict__ hprev, float* __restrict__ hout,
    bf16* __restrict__ hbf)
{
  const size_t idx = (size_t)blockIdx.x * 256 + threadIdx.x;   // B*DETER
  const size_t b = idx >> 11;
  const size_t j = idx & 2047;
  const size_t base = b * (size_t)GATES;
  const float ir = gi[base + j],          hr = gh[base + j];
  const float iz = gi[base + DETER + j],  hz = gh[base + DETER + j];
  const float in_ = gi[base + 2 * DETER + j], hn = gh[base + 2 * DETER + j];
  const float rr = 1.0f / (1.0f + __expf(-(ir + hr)));
  const float zz = 1.0f / (1.0f + __expf(-(iz + hz)));
  const float nn = tanhf(in_ + rr * hn);
  const float hp = hprev[idx];
  const float hnew = (1.0f - zz) * nn + zz * hp;
  hout[idx] = hnew;
  hbf[idx]  = (bf16)hnew;
}

// f32 -> bf16 with optional K padding (zeros)
__global__ __launch_bounds__(256) void convert_pad_kernel(
    const float* __restrict__ src, bf16* __restrict__ dst,
    int rows, int srcK, int dstK)
{
  const size_t idx = (size_t)blockIdx.x * 256 + threadIdx.x;
  const size_t total = (size_t)rows * dstK;
  if (idx >= total) return;
  const size_t rr = idx / dstK, cc = idx % dstK;
  dst[idx] = (cc < (size_t)srcK) ? (bf16)src[rr * (size_t)srcK + cc] : (bf16)0.0f;
}

// ---------------------------------------------------------------------------
// Workspace layout (bytes)
// ---------------------------------------------------------------------------
static constexpr size_t SZ_WMLP = 1024ull * INK   * 2;   //   2.2 MB
static constexpr size_t SZ_WIH  = 6144ull * 1024  * 2;   //  12.6 MB
static constexpr size_t SZ_WHH  = 6144ull * 2048  * 2;   //  25.2 MB
static constexpr size_t SZ_HBF  = 2048ull * 2048  * 2;   //   8.4 MB
static constexpr size_t SZ_FEAT = 65536ull * 1024 * 2;   // 134.2 MB
static constexpr size_t OFF_WMLP = 0;
static constexpr size_t OFF_WIH  = OFF_WMLP + SZ_WMLP;
static constexpr size_t OFF_WHH  = OFF_WIH + SZ_WIH;
static constexpr size_t OFF_HBF  = OFF_WHH + SZ_WHH;
static constexpr size_t OFF_FEAT = OFF_HBF + SZ_HBF;
static constexpr size_t OFF_BIG  = OFF_FEAT + SZ_FEAT;   // h_pre (268MB) reused as gi|gh

extern "C" void kernel_launch(void* const* d_in, const int* in_sizes, int n_in,
                              void* d_out, int out_size, void* d_ws, size_t ws_size,
                              hipStream_t stream) {
  const float* state_deter = (const float*)d_in[0];  // [1,B,DETER]
  const float* stoch       = (const float*)d_in[1];  // [T,B,1024]
  const float* actions     = (const float*)d_in[2];  // [T,B,48]
  const float* W_mlp       = (const float*)d_in[3];  // [1024,1072]
  const float* ln_gamma    = (const float*)d_in[4];
  const float* ln_beta     = (const float*)d_in[5];
  const float* W_ih        = (const float*)d_in[6];  // [6144,1024]
  const float* W_hh        = (const float*)d_in[7];  // [6144,2048]
  const float* b_ih        = (const float*)d_in[8];
  const float* b_hh        = (const float*)d_in[9];
  float* out = (float*)d_out;                        // [T,B,DETER]

  char* ws = (char*)d_ws;
  bf16*  Wmlp_bf = (bf16*)(ws + OFF_WMLP);
  bf16*  Wih_bf  = (bf16*)(ws + OFF_WIH);
  bf16*  Whh_bf  = (bf16*)(ws + OFF_WHH);
  bf16*  h_bf    = (bf16*)(ws + OFF_HBF);
  bf16*  feat_bf = (bf16*)(ws + OFF_FEAT);
  float* h_pre   = (float*)(ws + OFF_BIG);                       // [65536,1024] f32
  float* gi      = (float*)(ws + OFF_BIG);                       // [2048,6144] f32
  float* gh      = (float*)(ws + OFF_BIG + 2048ull * GATES * 4); // [2048,6144] f32

  // Weight / h0 conversion to bf16 (weights stay hot in the 192MB L2)
  auto cvt = [&](const float* s, bf16* dd, int rows, int sk, int dk) {
    const size_t tot = (size_t)rows * dk;
    convert_pad_kernel<<<(unsigned)((tot + 255) / 256), 256, 0, stream>>>(s, dd, rows, sk, dk);
  };
  cvt(W_mlp, Wmlp_bf, 1024, 1072, INK);
  cvt(W_ih,  Wih_bf,  6144, 1024, 1024);
  cvt(W_hh,  Whh_bf,  6144, 2048, 2048);
  cvt(state_deter, h_bf, 2048, 2048, 2048);

  // Phase 1: h_pre = concat(stoch, act) @ W_mlp^T   [65536 x 1024]
  wmma_gemm_concat_kernel<<<dim3(HID / BN, 65536 / BM), 256, 0, stream>>>(
      stoch, actions, Wmlp_bf, INK, h_pre, HID, INK);
  // LayerNorm + SiLU -> bf16 feat
  ln_silu_kernel<<<65536, 256, 0, stream>>>(h_pre, ln_gamma, ln_beta, feat_bf);

  // Phase 2+3: sequential GRU scan (weights L2-resident, GEMMs compute-bound)
  const dim3 ggrid(GATES / BN, BATCH / BM);
  for (int t = 0; t < TSTEPS; ++t) {
    const bf16* featT = feat_bf + (size_t)t * BATCH * HID;
    wmma_gemm_bias_kernel<<<ggrid, 256, 0, stream>>>(
        featT, HID, Wih_bf, HID, b_ih, gi, GATES, HID);
    wmma_gemm_bias_kernel<<<ggrid, 256, 0, stream>>>(
        h_bf, DETER, Whh_bf, DETER, b_hh, gh, GATES, DETER);
    const float* hprev = (t == 0) ? state_deter : out + (size_t)(t - 1) * BATCH * DETER;
    gru_pointwise_kernel<<<(BATCH * DETER) / 256, 256, 0, stream>>>(
        gi, gh, hprev, out + (size_t)t * BATCH * DETER, h_bf);
  }
}
